// CausalSelfAttention_75703093559686
// MI455X (gfx1250) — matrix operand order repaired and checked by reading, not compiled, not run on hardware
//
#include <hip/hip_runtime.h>
#include <hip/hip_bf16.h>
#include <cstddef>
#include <cstdint>

// ---------------------------------------------------------------------------
// Problem constants (fixed by the reference)
// ---------------------------------------------------------------------------
#define BATCH 2
#define S_LEN 1024
#define EMB   1024
#define NHEAD 16
#define DHEAD 64
#define SCALE 0.125f          // 1/sqrt(64)
#define LRATE 0.01f

typedef __bf16 bf16;
typedef __attribute__((ext_vector_type(16))) __bf16 v16bf;
typedef __attribute__((ext_vector_type(8)))  __bf16 v8bf;
typedef __attribute__((ext_vector_type(4)))  __bf16 v4bf;
typedef __attribute__((ext_vector_type(8)))  float  v8f;
typedef __attribute__((ext_vector_type(4)))  float  v4f;
typedef __attribute__((ext_vector_type(4)))  unsigned int u32x4;
typedef __attribute__((ext_vector_type(8)))  int i32x8;
typedef __attribute__((ext_vector_type(4)))  int i32x4;

#define WMMA_BF16(a, b, c) \
  __builtin_amdgcn_wmma_f32_16x16x32_bf16(false, (a), false, (b), (short)0, (c), false, false)

// Native f32 -> bf16 conversion (RNE); lets the compiler use packed cvt ops.
__device__ __forceinline__ bf16 f2bf(float x) { return (bf16)x; }

__device__ __forceinline__ v8f vz8() {
  v8f z = {0.f, 0.f, 0.f, 0.f, 0.f, 0.f, 0.f, 0.f};
  return z;
}

__device__ __forceinline__ v16bf mergebf(v8bf lo, v8bf hi) {
  return __builtin_shufflevector(lo, hi, 0,1,2,3,4,5,6,7,8,9,10,11,12,13,14,15);
}

// A-fragment (16x32, 16-bit): lane m = lane&15; half=lane>>4 selects
// K groups: e0..7 -> K = 8*half + (0..7); e8..15 -> K = 16 + 8*half + (0..7)
__device__ __forceinline__ v16bf ldsFragA(const bf16* base, int stride, int row,
                                          int kOff, int half) {
  const bf16* p = base + row * stride + kOff + 8 * half;
  v8bf lo = *(const v8bf*)p;
  v8bf hi = *(const v8bf*)(p + 16);
  return mergebf(lo, hi);
}

// B-fragment (32x16, 16-bit): lane n = lane&15; half=lane>>4 selects
// K groups: e0..7 -> K = 8*half + (0..7); e8..15 -> K = 16 + 8*half + (0..7)
__device__ __forceinline__ v16bf ldsFragB(const bf16* base, int stride, int n,
                                          int kOff, int half) {
  const bf16* p = base + n * stride + kOff + 8 * half;
  v8bf lo = *(const v8bf*)p;
  v8bf hi = *(const v8bf*)(p + 16);
  return mergebf(lo, hi);
}

// Build an A-fragment directly from a global fp32 row chunk (32 wide).
__device__ __forceinline__ v16bf gblFragA(const float* rowPtr, int half) {
  const float* p0 = rowPtr + 8 * half;
  const float* p1 = rowPtr + 16 + 8 * half;
  v8bf lo, hi;
#pragma unroll
  for (int i = 0; i < 8; i++) { lo[i] = f2bf(p0[i]); hi[i] = f2bf(p1[i]); }
  return mergebf(lo, hi);
}

// ---------------------------------------------------------------------------
// Generic NT GEMM:  C[M,N] = alpha * A[M,K] @ W[N,K]^T  (+ beta * Add)
// A is fp32 (converted while staging); W is a pre-converted bf16 buffer whose
// 128x32 tile is DMA'd into LDS by the Tensor Data Mover each K-step.
// 128x128 block tile, K-step 32, 256 threads = 8 waves (2x4), wave tile 64x32.
// ---------------------------------------------------------------------------
__global__ __launch_bounds__(256) void gemm_nt_kernel(
    const float* __restrict__ A, const bf16* __restrict__ W,
    const float* __restrict__ Add, float* __restrict__ C,
    int M, int N, int K, float alpha, float beta) {
  __shared__ bf16 As[128 * 40];
  __shared__ bf16 Bs[128 * 40];   // TDM writes 64B rows + 16B pad -> stride 40

  const int tid = threadIdx.x;
  const int lane = tid & 31, wave = tid >> 5;
  const int half = lane >> 4, l16 = lane & 15;
  const int waveM = wave & 1, waveN = wave >> 1;   // 2 x 4 wave grid
  const int blockM = blockIdx.y * 128, blockN = blockIdx.x * 128;

  v8f acc[4][2];
#pragma unroll
  for (int mt = 0; mt < 4; mt++)
#pragma unroll
    for (int nt = 0; nt < 2; nt++) acc[mt][nt] = vz8();

  for (int kk = 0; kk < K; kk += 32) {
    // ---- stage A tile (fp32 -> bf16) ----
#pragma unroll
    for (int j = 0; j < 4; j++) {
      int i4 = tid + j * 256;          // 1024 float4 slots: 128 rows x 8
      int r = i4 >> 3, c4 = i4 & 7;
      const float* pa = A + (size_t)(blockM + r) * K + kk + c4 * 4;
      if (kk + 32 < K) __builtin_prefetch(pa + 32, 0, 3);
      v4f va = *(const v4f*)pa;
      v4bf oa;
#pragma unroll
      for (int x = 0; x < 4; x++) oa[x] = f2bf(va[x]);
      *(v4bf*)&As[r * 40 + c4 * 4] = oa;
    }

    // ---- stage W tile via Tensor Data Mover (wave 0 issues one descriptor) --
    if (wave == 0) {
      unsigned ldsB = (unsigned)(uintptr_t)&Bs[0];
      unsigned long long ga =
          (unsigned long long)(uintptr_t)(W + (size_t)blockN * K + kk);
      u32x4 g0;
      g0[0] = 1u;                                        // count=1 (valid user D#)
      g0[1] = ldsB;                                      // lds_addr
      g0[2] = (unsigned)(ga & 0xFFFFFFFFull);            // global_addr[31:0]
      g0[3] = (unsigned)((ga >> 32) & 0x01FFFFFFull) | 0x80000000u; // [56:32]|type=2
      i32x8 g1;
      // data_size=1(2B) | pad_enable | pad_interval=3(16 DW) | pad_amount=3(4 DW)
      g1[0] = (int)((1u << 16) | (1u << 20) | (3u << 22) | (3u << 25));
      unsigned td0 = (unsigned)(K - kk);                 // tensor_dim0
      unsigned td1 = (unsigned)(N - blockN);             // tensor_dim1
      g1[1] = (int)(td0 << 16);                          // tensor_dim0[15:0]
      g1[2] = (int)((td0 >> 16) | (td1 << 16));          // dim0 hi | dim1 lo
      g1[3] = (int)(((td1 >> 16) & 0xFFFFu) | (32u << 16)); // dim1 hi | tile_dim0=32
      g1[4] = 128;                                       // tile_dim1=128 rows
      g1[5] = K;                                         // tensor_dim0_stride
      g1[6] = 0;
      g1[7] = 0;
      i32x4 g2 = {0, 0, 0, 0};                           // 2D: groups 2/3 unused
      i32x4 g3 = {0, 0, 0, 0};
      i32x8 g4 = {0, 0, 0, 0, 0, 0, 0, 0};               // extra group (clang-23 form)
      __builtin_amdgcn_tensor_load_to_lds(g0, g1, g2, g3, g4, 0);
      __builtin_amdgcn_s_wait_tensorcnt(0);
    }
    __syncthreads();

    v16bf af[4], bfv[2];
#pragma unroll
    for (int mt = 0; mt < 4; mt++)
      af[mt] = ldsFragA(As, 40, waveM * 64 + mt * 16 + l16, 0, half);
#pragma unroll
    for (int nt = 0; nt < 2; nt++)
      bfv[nt] = ldsFragB(Bs, 40, waveN * 32 + nt * 16 + l16, 0, half);
#pragma unroll
    for (int mt = 0; mt < 4; mt++)
#pragma unroll
      for (int nt = 0; nt < 2; nt++)
        acc[mt][nt] = WMMA_BF16(af[mt], bfv[nt], acc[mt][nt]);
    __syncthreads();
  }

#pragma unroll
  for (int mt = 0; mt < 4; mt++)
#pragma unroll
    for (int nt = 0; nt < 2; nt++)
#pragma unroll
      for (int i = 0; i < 8; i++) {
        int m = blockM + waveM * 64 + mt * 16 + i + 8 * half;
        int n = blockN + waveN * 32 + nt * 16 + l16;
        size_t idx = (size_t)m * N + n;
        float v = alpha * acc[mt][nt][i];
        if (Add) v += beta * Add[idx];
        C[idx] = v;
      }
}

// ---------------------------------------------------------------------------
// fp32 -> bf16 buffer conversion, and fp32 -> bf16 transpose (E x E)
// ---------------------------------------------------------------------------
__global__ void cvt_bf16_kernel(const float* __restrict__ in,
                                bf16* __restrict__ out, size_t n) {
  size_t i = (size_t)blockIdx.x * blockDim.x + threadIdx.x;
  if (i < n) out[i] = f2bf(in[i]);
}

__global__ void transpose_bf16_kernel(const float* __restrict__ in,
                                      bf16* __restrict__ out, int n) {
  __shared__ float t[32][33];
  int bx = blockIdx.x * 32, by = blockIdx.y * 32;
  int x = threadIdx.x, y = threadIdx.y;   // 32 x 8
#pragma unroll
  for (int j = 0; j < 32; j += 8) t[y + j][x] = in[(size_t)(by + y + j) * n + bx + x];
  __syncthreads();
#pragma unroll
  for (int j = 0; j < 32; j += 8)
    out[(size_t)(bx + y + j) * n + by + x] = f2bf(t[x][y + j]);
}

// ---------------------------------------------------------------------------
// Flash attention forward (causal). Grid: (S/64, NHEAD, BATCH), 128 threads.
// Each wave owns 16 q rows; K/V streamed in 32-row chunks through LDS.
// ---------------------------------------------------------------------------
__global__ __launch_bounds__(128) void attn_fwd_kernel(
    const float* __restrict__ qg, const float* __restrict__ kg,
    const float* __restrict__ vg, float* __restrict__ og,
    float* __restrict__ lse) {
  __shared__ bf16 Ks[32 * 72];
  __shared__ bf16 Vt[64 * 40];
  __shared__ bf16 Pscr[4][16 * 40];

  const int tid = threadIdx.x, lane = tid & 31, wq = tid >> 5;
  const int half = lane >> 4, l16 = lane & 15;
  const int b = blockIdx.z, h = blockIdx.y;
  const int qBlk = blockIdx.x * 64;
  const int qRow0 = qBlk + wq * 16;
  const size_t baseBH = ((size_t)b * S_LEN) * EMB + h * DHEAD;
  const size_t lb = ((size_t)b * NHEAD + h) * S_LEN;

  v16bf aQ[2];
#pragma unroll
  for (int dc = 0; dc < 2; dc++)
    aQ[dc] = gblFragA(qg + baseBH + (size_t)(qRow0 + l16) * EMB + dc * 32, half);

  float mrow[8], lrow[8];
  v8f accO[4];
#pragma unroll
  for (int i = 0; i < 8; i++) { mrow[i] = -3.0e38f; lrow[i] = 0.f; }
#pragma unroll
  for (int dt = 0; dt < 4; dt++) accO[dt] = vz8();

  for (int kb = 0; kb < qBlk + 64; kb += 32) {
#pragma unroll
    for (int j = 0; j < 4; j++) {
      int i4 = tid + j * 128;           // 512 float4 slots: 32 rows x 16
      int r = i4 >> 4, c4 = i4 & 15;
      v4f kv = *(const v4f*)(kg + baseBH + (size_t)(kb + r) * EMB + c4 * 4);
      v4bf k4;
#pragma unroll
      for (int x = 0; x < 4; x++) k4[x] = f2bf(kv[x]);
      *(v4bf*)&Ks[r * 72 + c4 * 4] = k4;
      v4f vv = *(const v4f*)(vg + baseBH + (size_t)(kb + r) * EMB + c4 * 4);
#pragma unroll
      for (int x = 0; x < 4; x++) Vt[(c4 * 4 + x) * 40 + r] = f2bf(vv[x]);
    }
    __syncthreads();

    v8f sc[2];
    sc[0] = vz8(); sc[1] = vz8();
#pragma unroll
    for (int dc = 0; dc < 2; dc++)
#pragma unroll
      for (int nt = 0; nt < 2; nt++) {
        v16bf bk = ldsFragB(Ks, 72, nt * 16 + l16, dc * 32, half);
        sc[nt] = WMMA_BF16(aQ[dc], bk, sc[nt]);
      }

#pragma unroll
    for (int i = 0; i < 8; i++) {
      int qgI = qRow0 + i + 8 * half;
      float s0 = sc[0][i] * SCALE, s1 = sc[1][i] * SCALE;
      if (kb + l16 > qgI)      s0 = -3.0e38f;
      if (kb + 16 + l16 > qgI) s1 = -3.0e38f;
      float rm = fmaxf(s0, s1);
      rm = fmaxf(rm, __shfl_xor(rm, 1));
      rm = fmaxf(rm, __shfl_xor(rm, 2));
      rm = fmaxf(rm, __shfl_xor(rm, 4));
      rm = fmaxf(rm, __shfl_xor(rm, 8));
      float mnew = fmaxf(mrow[i], rm);
      float p0 = __expf(s0 - mnew), p1 = __expf(s1 - mnew);
      float rs = p0 + p1;
      rs += __shfl_xor(rs, 1);
      rs += __shfl_xor(rs, 2);
      rs += __shfl_xor(rs, 4);
      rs += __shfl_xor(rs, 8);
      float corr = __expf(mrow[i] - mnew);
      lrow[i] = lrow[i] * corr + rs;
      mrow[i] = mnew;
#pragma unroll
      for (int dt = 0; dt < 4; dt++) accO[dt][i] *= corr;
      Pscr[wq][(i + 8 * half) * 40 + l16] = f2bf(p0);
      Pscr[wq][(i + 8 * half) * 40 + 16 + l16] = f2bf(p1);
    }

    v16bf aP = ldsFragA(&Pscr[wq][0], 40, l16, 0, half);
#pragma unroll
    for (int dt = 0; dt < 4; dt++) {
      v16bf bv = ldsFragB(Vt, 40, dt * 16 + l16, 0, half);
      accO[dt] = WMMA_BF16(aP, bv, accO[dt]);
    }
    __syncthreads();
  }

#pragma unroll
  for (int i = 0; i < 8; i++) {
    int qgI = qRow0 + i + 8 * half;
    float inv = 1.0f / lrow[i];
#pragma unroll
    for (int dt = 0; dt < 4; dt++)
      og[baseBH + (size_t)qgI * EMB + dt * 16 + l16] = accO[dt][i] * inv;
    if (l16 == 0) lse[lb + qgI] = mrow[i] + __logf(lrow[i]);
  }
}

// ---------------------------------------------------------------------------
// Flash backward, dK/dV pass. Grid: (S/64 k-tiles, NHEAD, BATCH), 128 thr.
// ---------------------------------------------------------------------------
__global__ __launch_bounds__(128) void attn_bwd_kv_kernel(
    const float* __restrict__ qg, const float* __restrict__ kg,
    const float* __restrict__ vg, const float* __restrict__ dOg,
    const float* __restrict__ lse, const float* __restrict__ Dsum,
    float* __restrict__ dkg, float* __restrict__ dvg) {
  __shared__ bf16 Qr[32 * 72];
  __shared__ bf16 Qt[64 * 40];
  __shared__ bf16 dOr[32 * 72];
  __shared__ bf16 dOt[64 * 40];
  __shared__ bf16 Pscr[4][16 * 40];
  __shared__ float lseS[32], DsS[32];

  const int tid = threadIdx.x, lane = tid & 31, wq = tid >> 5;
  const int half = lane >> 4, l16 = lane & 15;
  const int b = blockIdx.z, h = blockIdx.y;
  const int kBlk = blockIdx.x * 64;
  const int kRow0 = kBlk + wq * 16;
  const size_t baseBH = ((size_t)b * S_LEN) * EMB + h * DHEAD;
  const size_t lb = ((size_t)b * NHEAD + h) * S_LEN;

  v16bf aK[2], aV[2];
#pragma unroll
  for (int dc = 0; dc < 2; dc++) {
    aK[dc] = gblFragA(kg + baseBH + (size_t)(kRow0 + l16) * EMB + dc * 32, half);
    aV[dc] = gblFragA(vg + baseBH + (size_t)(kRow0 + l16) * EMB + dc * 32, half);
  }

  v8f accDV[4], accDK[4];
#pragma unroll
  for (int dt = 0; dt < 4; dt++) { accDV[dt] = vz8(); accDK[dt] = vz8(); }

  for (int qc = kBlk; qc < S_LEN; qc += 32) {
#pragma unroll
    for (int j = 0; j < 4; j++) {
      int i4 = tid + j * 128;
      int r = i4 >> 4, c4 = i4 & 15;
      v4f qv = *(const v4f*)(qg + baseBH + (size_t)(qc + r) * EMB + c4 * 4);
      v4bf q4;
#pragma unroll
      for (int x = 0; x < 4; x++) q4[x] = f2bf(qv[x]);
      *(v4bf*)&Qr[r * 72 + c4 * 4] = q4;
#pragma unroll
      for (int x = 0; x < 4; x++) Qt[(c4 * 4 + x) * 40 + r] = q4[x];
      v4f dv4 = *(const v4f*)(dOg + baseBH + (size_t)(qc + r) * EMB + c4 * 4);
      v4bf d4;
#pragma unroll
      for (int x = 0; x < 4; x++) d4[x] = f2bf(dv4[x]);
      *(v4bf*)&dOr[r * 72 + c4 * 4] = d4;
#pragma unroll
      for (int x = 0; x < 4; x++) dOt[(c4 * 4 + x) * 40 + r] = d4[x];
    }
    if (tid < 32) { lseS[tid] = lse[lb + qc + tid]; DsS[tid] = Dsum[lb + qc + tid]; }
    __syncthreads();

    // St[kr][q] = K . Q^T
    v8f st[2];
    st[0] = vz8(); st[1] = vz8();
#pragma unroll
    for (int dc = 0; dc < 2; dc++)
#pragma unroll
      for (int nt = 0; nt < 2; nt++) {
        v16bf bq = ldsFragB(Qr, 72, nt * 16 + l16, dc * 32, half);
        st[nt] = WMMA_BF16(aK[dc], bq, st[nt]);
      }

    float lse0 = lseS[l16], lse1 = lseS[16 + l16];
    float Ds0 = DsS[l16], Ds1 = DsS[16 + l16];
    v8f pt0, pt1;
#pragma unroll
    for (int i = 0; i < 8; i++) {
      int krg = kRow0 + i + 8 * half;
      float p0 = (qc + l16 >= krg)      ? __expf(st[0][i] * SCALE - lse0) : 0.f;
      float p1 = (qc + 16 + l16 >= krg) ? __expf(st[1][i] * SCALE - lse1) : 0.f;
      pt0[i] = p0; pt1[i] = p1;
      Pscr[wq][(i + 8 * half) * 40 + l16] = f2bf(p0);
      Pscr[wq][(i + 8 * half) * 40 + 16 + l16] = f2bf(p1);
    }

    v16bf aPt = ldsFragA(&Pscr[wq][0], 40, l16, 0, half);
#pragma unroll
    for (int dt = 0; dt < 4; dt++) {
      v16bf bdo = ldsFragB(dOt, 40, dt * 16 + l16, 0, half);
      accDV[dt] = WMMA_BF16(aPt, bdo, accDV[dt]);
    }

    // dPt[kr][q] = V . dO^T
    v8f dpt[2];
    dpt[0] = vz8(); dpt[1] = vz8();
#pragma unroll
    for (int dc = 0; dc < 2; dc++)
#pragma unroll
      for (int nt = 0; nt < 2; nt++) {
        v16bf bd = ldsFragB(dOr, 72, nt * 16 + l16, dc * 32, half);
        dpt[nt] = WMMA_BF16(aV[dc], bd, dpt[nt]);
      }

#pragma unroll
    for (int i = 0; i < 8; i++) {
      float ds0 = pt0[i] * (dpt[0][i] - Ds0);
      float ds1 = pt1[i] * (dpt[1][i] - Ds1);
      Pscr[wq][(i + 8 * half) * 40 + l16] = f2bf(ds0);
      Pscr[wq][(i + 8 * half) * 40 + 16 + l16] = f2bf(ds1);
    }
    v16bf aDS = ldsFragA(&Pscr[wq][0], 40, l16, 0, half);
#pragma unroll
    for (int dt = 0; dt < 4; dt++) {
      v16bf bqt = ldsFragB(Qt, 40, dt * 16 + l16, 0, half);
      accDK[dt] = WMMA_BF16(aDS, bqt, accDK[dt]);
    }
    __syncthreads();
  }

#pragma unroll
  for (int i = 0; i < 8; i++) {
    int krg = kRow0 + i + 8 * half;
#pragma unroll
    for (int dt = 0; dt < 4; dt++) {
      dvg[baseBH + (size_t)krg * EMB + dt * 16 + l16] = accDV[dt][i];
      dkg[baseBH + (size_t)krg * EMB + dt * 16 + l16] = accDK[dt][i] * SCALE;
    }
  }
}

// ---------------------------------------------------------------------------
// Flash backward, dQ pass. Grid: (S/64 q-tiles, NHEAD, BATCH), 128 thr.
// ---------------------------------------------------------------------------
__global__ __launch_bounds__(128) void attn_bwd_q_kernel(
    const float* __restrict__ qg, const float* __restrict__ kg,
    const float* __restrict__ vg, const float* __restrict__ dOg,
    const float* __restrict__ lse, const float* __restrict__ Dsum,
    float* __restrict__ dqg) {
  __shared__ bf16 Kr[32 * 72];
  __shared__ bf16 Kt[64 * 40];
  __shared__ bf16 Vr[32 * 72];
  __shared__ bf16 Pscr[4][16 * 40];

  const int tid = threadIdx.x, lane = tid & 31, wq = tid >> 5;
  const int half = lane >> 4, l16 = lane & 15;
  const int b = blockIdx.z, h = blockIdx.y;
  const int qBlk = blockIdx.x * 64;
  const int qRow0 = qBlk + wq * 16;
  const size_t baseBH = ((size_t)b * S_LEN) * EMB + h * DHEAD;
  const size_t lb = ((size_t)b * NHEAD + h) * S_LEN;

  v16bf aQ[2], adO[2];
#pragma unroll
  for (int dc = 0; dc < 2; dc++) {
    aQ[dc]  = gblFragA(qg + baseBH + (size_t)(qRow0 + l16) * EMB + dc * 32, half);
    adO[dc] = gblFragA(dOg + baseBH + (size_t)(qRow0 + l16) * EMB + dc * 32, half);
  }
  float lseR[8], DR[8];
#pragma unroll
  for (int i = 0; i < 8; i++) {
    int qgI = qRow0 + i + 8 * half;
    lseR[i] = lse[lb + qgI];
    DR[i] = Dsum[lb + qgI];
  }

  v8f accDQ[4];
#pragma unroll
  for (int dt = 0; dt < 4; dt++) accDQ[dt] = vz8();

  for (int kc = 0; kc < qBlk + 64; kc += 32) {
#pragma unroll
    for (int j = 0; j < 4; j++) {
      int i4 = tid + j * 128;
      int r = i4 >> 4, c4 = i4 & 15;
      v4f kv = *(const v4f*)(kg + baseBH + (size_t)(kc + r) * EMB + c4 * 4);
      v4bf k4;
#pragma unroll
      for (int x = 0; x < 4; x++) k4[x] = f2bf(kv[x]);
      *(v4bf*)&Kr[r * 72 + c4 * 4] = k4;
#pragma unroll
      for (int x = 0; x < 4; x++) Kt[(c4 * 4 + x) * 40 + r] = k4[x];
      v4f vv = *(const v4f*)(vg + baseBH + (size_t)(kc + r) * EMB + c4 * 4);
      v4bf v4b;
#pragma unroll
      for (int x = 0; x < 4; x++) v4b[x] = f2bf(vv[x]);
      *(v4bf*)&Vr[r * 72 + c4 * 4] = v4b;
    }
    __syncthreads();

    v8f sc[2], dp[2];
    sc[0] = vz8(); sc[1] = vz8(); dp[0] = vz8(); dp[1] = vz8();
#pragma unroll
    for (int dc = 0; dc < 2; dc++)
#pragma unroll
      for (int nt = 0; nt < 2; nt++) {
        v16bf bk = ldsFragB(Kr, 72, nt * 16 + l16, dc * 32, half);
        sc[nt] = WMMA_BF16(aQ[dc], bk, sc[nt]);
        v16bf bv = ldsFragB(Vr, 72, nt * 16 + l16, dc * 32, half);
        dp[nt] = WMMA_BF16(adO[dc], bv, dp[nt]);
      }

#pragma unroll
    for (int i = 0; i < 8; i++) {
      int qgI = qRow0 + i + 8 * half;
      float p0 = (kc + l16 <= qgI)      ? __expf(sc[0][i] * SCALE - lseR[i]) : 0.f;
      float p1 = (kc + 16 + l16 <= qgI) ? __expf(sc[1][i] * SCALE - lseR[i]) : 0.f;
      float ds0 = p0 * (dp[0][i] - DR[i]);
      float ds1 = p1 * (dp[1][i] - DR[i]);
      Pscr[wq][(i + 8 * half) * 40 + l16] = f2bf(ds0);
      Pscr[wq][(i + 8 * half) * 40 + 16 + l16] = f2bf(ds1);
    }
    v16bf aDS = ldsFragA(&Pscr[wq][0], 40, l16, 0, half);
#pragma unroll
    for (int dt = 0; dt < 4; dt++) {
      v16bf bkt = ldsFragB(Kt, 40, dt * 16 + l16, 0, half);
      accDQ[dt] = WMMA_BF16(aDS, bkt, accDQ[dt]);
    }
    __syncthreads();
  }

#pragma unroll
  for (int i = 0; i < 8; i++) {
    int qgI = qRow0 + i + 8 * half;
#pragma unroll
    for (int dt = 0; dt < 4; dt++)
      dqg[baseBH + (size_t)qgI * EMB + dt * 16 + l16] = accDQ[dt][i] * SCALE;
  }
}

// ---------------------------------------------------------------------------
// Small helper kernels
// ---------------------------------------------------------------------------
__global__ void dsum_kernel(const float* __restrict__ dO,
                            const float* __restrict__ o,
                            float* __restrict__ Dsum) {
  int idx = blockIdx.x * blockDim.x + threadIdx.x;  // [0, B*H*S)
  if (idx >= BATCH * NHEAD * S_LEN) return;
  int s = idx % S_LEN;
  int h = (idx / S_LEN) % NHEAD;
  int b = idx / (S_LEN * NHEAD);
  size_t base = ((size_t)b * S_LEN + s) * EMB + h * DHEAD;
  float acc = 0.f;
#pragma unroll 8
  for (int d = 0; d < DHEAD; d++) acc += dO[base + d] * o[base + d];
  Dsum[idx] = acc;
}

__global__ void ew_sub_kernel(const float* __restrict__ a,
                              const float* __restrict__ b,
                              float* __restrict__ c, size_t n) {
  size_t i = (size_t)blockIdx.x * blockDim.x + threadIdx.x;
  if (i < n) c[i] = a[i] - b[i];
}

__global__ void ew_zero_kernel(float* __restrict__ a, size_t n) {
  size_t i = (size_t)blockIdx.x * blockDim.x + threadIdx.x;
  if (i < n) a[i] = 0.f;
}

// ---------------------------------------------------------------------------
// Host-side launch
// ---------------------------------------------------------------------------
extern "C" void kernel_launch(void* const* d_in, const int* in_sizes, int n_in,
                              void* d_out, int out_size, void* d_ws, size_t ws_size,
                              hipStream_t stream) {
  (void)in_sizes; (void)n_in; (void)out_size; (void)ws_size;
  const float* t  = (const float*)d_in[0];   // fc1_x [B,S,E]
  const float* Wq = (const float*)d_in[1];
  const float* Wk = (const float*)d_in[2];
  const float* Wv = (const float*)d_in[3];
  const float* Wo = (const float*)d_in[4];

  const size_t ME = (size_t)BATCH * S_LEN * EMB;   // 2,097,152
  const size_t EE = (size_t)EMB * EMB;

  float* w = (float*)d_ws;
  float* x1  = w; w += ME;   // stage-1 latent -> att_score (stage-2 target)
  float* x2  = w; w += ME;   // stage-2 latent (result)
  float* rb  = w; w += ME;   // residual / dO
  float* qB  = w; w += ME;
  float* kB  = w; w += ME;
  float* vB  = w; w += ME;
  float* oB  = w; w += ME;
  float* dqB = w; w += ME;
  float* dkB = w; w += ME;
  float* dvB = w; w += ME;
  float* lseB = w; w += (size_t)BATCH * NHEAD * S_LEN;
  float* DsB  = w; w += (size_t)BATCH * NHEAD * S_LEN;
  // bf16 weight copies (both orientations) for the TDM-fed GEMM B operand
  bf16* wb = (bf16*)w;
  bf16* Wo_b  = wb; wb += EE;
  bf16* Wq_b  = wb; wb += EE;
  bf16* Wk_b  = wb; wb += EE;
  bf16* Wv_b  = wb; wb += EE;
  bf16* WoT_b = wb; wb += EE;
  bf16* WqT_b = wb; wb += EE;
  bf16* WkT_b = wb; wb += EE;
  bf16* WvT_b = wb; wb += EE;

  const int M = BATCH * S_LEN, N = EMB, K = EMB;
  dim3 gT(EMB / 32, EMB / 32), bT(32, 8);
  dim3 gG(N / 128, M / 128), bG(256);
  dim3 gA(S_LEN / 64, NHEAD, BATCH), bA(128);
  const int ewBlocks = (int)((ME + 255) / 256);
  const int cvBlocks = (int)((EE + 255) / 256);
  const int dsBlocks = (BATCH * NHEAD * S_LEN + 255) / 256;

  // Weight preparation: bf16 copies + bf16 transposes (every GEMM is NT).
  cvt_bf16_kernel<<<cvBlocks, 256, 0, stream>>>(Wo, Wo_b, EE);
  cvt_bf16_kernel<<<cvBlocks, 256, 0, stream>>>(Wq, Wq_b, EE);
  cvt_bf16_kernel<<<cvBlocks, 256, 0, stream>>>(Wk, Wk_b, EE);
  cvt_bf16_kernel<<<cvBlocks, 256, 0, stream>>>(Wv, Wv_b, EE);
  transpose_bf16_kernel<<<gT, bT, 0, stream>>>(Wo, WoT_b, EMB);
  transpose_bf16_kernel<<<gT, bT, 0, stream>>>(Wq, WqT_b, EMB);
  transpose_bf16_kernel<<<gT, bT, 0, stream>>>(Wk, WkT_b, EMB);
  transpose_bf16_kernel<<<gT, bT, 0, stream>>>(Wv, WvT_b, EMB);

  // -------- Stage 1: x1 <- x1 + LR * (t - x1 Wo^T) Wo, x1_0 = 0 --------
  // step 1 collapses to x1 = LR * (t @ Wo) = LR * t @ WoT^T
  gemm_nt_kernel<<<gG, bG, 0, stream>>>(t, WoT_b, nullptr, x1, M, N, K, LRATE, 0.f);
  for (int it = 0; it < 2; it++) {
    gemm_nt_kernel<<<gG, bG, 0, stream>>>(x1, Wo_b, t, rb, M, N, K, -1.f, 1.f);    // r = t - x1 Wo^T
    gemm_nt_kernel<<<gG, bG, 0, stream>>>(rb, WoT_b, x1, x1, M, N, K, LRATE, 1.f); // x1 += LR r Wo
  }

  // -------- Stage 2: x2 <- x2 + LR * MHA_vjp(x2, x1 - MHA(x2)) --------
  ew_zero_kernel<<<ewBlocks, 256, 0, stream>>>(x2, ME);
  for (int it = 0; it < 3; it++) {
    gemm_nt_kernel<<<gG, bG, 0, stream>>>(x2, Wq_b, nullptr, qB, M, N, K, 1.f, 0.f);
    gemm_nt_kernel<<<gG, bG, 0, stream>>>(x2, Wk_b, nullptr, kB, M, N, K, 1.f, 0.f);
    gemm_nt_kernel<<<gG, bG, 0, stream>>>(x2, Wv_b, nullptr, vB, M, N, K, 1.f, 0.f);
    attn_fwd_kernel<<<gA, bA, 0, stream>>>(qB, kB, vB, oB, lseB);
    ew_sub_kernel<<<ewBlocks, 256, 0, stream>>>(x1, oB, rb, ME);        // dO = t2 - o
    dsum_kernel<<<dsBlocks, 256, 0, stream>>>(rb, oB, DsB);
    attn_bwd_kv_kernel<<<gA, bA, 0, stream>>>(qB, kB, vB, rb, lseB, DsB, dkB, dvB);
    attn_bwd_q_kernel<<<gA, bA, 0, stream>>>(qB, kB, vB, rb, lseB, DsB, dqB);
    gemm_nt_kernel<<<gG, bG, 0, stream>>>(dqB, WqT_b, x2, x2, M, N, K, LRATE, 1.f);
    gemm_nt_kernel<<<gG, bG, 0, stream>>>(dkB, WkT_b, x2, x2, M, N, K, LRATE, 1.f);
    gemm_nt_kernel<<<gG, bG, 0, stream>>>(dvB, WvT_b, x2, x2, M, N, K, LRATE, 1.f);
  }

  (void)hipMemcpyAsync(d_out, x2, ME * sizeof(float), hipMemcpyDeviceToDevice, stream);
}
